// PRED_52192442581397
// MI455X (gfx1250) — matrix operand order repaired and checked by reading, not compiled, not run on hardware
//
#include <hip/hip_runtime.h>
#include <hip/hip_bf16.h>

// ---------------------------------------------------------------------------
// Types for CDNA5 WMMA (wave32): v_wmma_f32_16x16x32_bf16  + TDM descriptors
// ---------------------------------------------------------------------------
typedef __attribute__((ext_vector_type(16))) __bf16          v16bf;
typedef __attribute__((ext_vector_type(8)))  float           v8f;
typedef __attribute__((ext_vector_type(8)))  unsigned short  u16x8;
typedef __attribute__((ext_vector_type(16))) unsigned short  u16x16;
typedef __attribute__((ext_vector_type(4)))  unsigned int    v4u;
typedef __attribute__((ext_vector_type(8)))  int             v8i;
typedef __attribute__((ext_vector_type(4)))  int             v4i;

#if defined(__has_builtin)
#  if __has_builtin(__builtin_amdgcn_tensor_load_to_lds)
#    define HAVE_TDM 1
#  endif
#endif
#ifndef HAVE_TDM
#  define HAVE_TDM 0
#endif

#define EPS 1e-5f

__device__ __forceinline__ unsigned short f2bf(float f) {
  unsigned int u = __float_as_uint(f);
  unsigned int r = (u + 0x7FFFu + ((u >> 16) & 1u)) >> 16;  // round-nearest-even
  return (unsigned short)r;
}
__device__ __forceinline__ float bf2f(unsigned short h) {
  return __uint_as_float(((unsigned int)h) << 16);
}

// ---------------------------------------------------------------------------
// K0: hypernet weight gather. onehot @ W.T + b  ==  W[:, action[b]] + b.
// Produces per-sample bf16 weights laid out [B][N][Kpad], K = Cin*25,
// k = ci*25 + ky*5 + kx (matches reshape (Cout, Cin, 5, 5)), zero-padded.
// ---------------------------------------------------------------------------
__global__ void gather_w_kernel(const float* __restrict__ w,
                                const float* __restrict__ bias,
                                const int*   __restrict__ action,
                                unsigned short* __restrict__ outw,
                                int N, int K, int Kpad) {
  size_t total = (size_t)16 * N * Kpad;
  for (size_t i = (size_t)blockIdx.x * blockDim.x + threadIdx.x; i < total;
       i += (size_t)gridDim.x * blockDim.x) {
    int b   = (int)(i / ((size_t)N * Kpad));
    int rem = (int)(i % ((size_t)N * Kpad));
    int n = rem / Kpad, k = rem - n * Kpad;
    unsigned short v = 0;
    if (k < K) {
      int od = n * K + k;
      float f = w[(size_t)od * 6 + action[b]] + bias[od];
      v = f2bf(f);
    }
    outw[i] = v;
  }
}

// ---------------------------------------------------------------------------
// K1: per-(sample,channel) mean / inv-std over H*W (train-mode BN, biased var)
// ---------------------------------------------------------------------------
__global__ __launch_bounds__(256) void bn_stats_kernel(
    const float* __restrict__ raw, float* __restrict__ stats, int HW) {
  int bc = blockIdx.x;
  const float* p = raw + (size_t)bc * HW;
  float s = 0.f, s2 = 0.f;
  for (int i = threadIdx.x; i < HW; i += 256) {
    float v = p[i];
    s += v; s2 += v * v;
  }
  __shared__ float rs[256], rq[256];
  rs[threadIdx.x] = s; rq[threadIdx.x] = s2;
  __syncthreads();
  for (int o = 128; o > 0; o >>= 1) {
    if (threadIdx.x < o) {
      rs[threadIdx.x] += rs[threadIdx.x + o];
      rq[threadIdx.x] += rq[threadIdx.x + o];
    }
    __syncthreads();
  }
  if (threadIdx.x == 0) {
    float m = rs[0] / (float)HW;
    float var = rq[0] / (float)HW - m * m;
    stats[2 * bc]     = m;
    stats[2 * bc + 1] = rsqrtf(fmaxf(var, 0.f) + EPS);
  }
}

// ---------------------------------------------------------------------------
// K2: BN apply (+optional ReLU), fp32 -> bf16 activation
// ---------------------------------------------------------------------------
__global__ void bn_apply_kernel(const float* __restrict__ raw,
                                unsigned short* __restrict__ act,
                                const float* __restrict__ stats,
                                const float* __restrict__ scale,
                                const float* __restrict__ bias,
                                int CH, int HW, int relu, size_t total) {
  for (size_t i = (size_t)blockIdx.x * blockDim.x + threadIdx.x; i < total;
       i += (size_t)gridDim.x * blockDim.x) {
    size_t bc = i / (size_t)HW;          // b*CH + c
    int c = (int)(bc % CH);
    float v = (raw[i] - stats[2 * bc]) * stats[2 * bc + 1] * scale[c] + bias[c];
    if (relu) v = fmaxf(v, 0.f);
    act[i] = f2bf(v);
  }
}

// ---------------------------------------------------------------------------
// K3: layer-2 special: avg-pool 2x2 of raw conv output, then BN (affine
// commutes with the linear pool; stats computed on the un-pooled map).
// ---------------------------------------------------------------------------
__global__ void pool_bn_kernel(const float* __restrict__ raw,
                               unsigned short* __restrict__ act,
                               const float* __restrict__ stats,
                               const float* __restrict__ scale,
                               const float* __restrict__ bias) {
  size_t total = (size_t)16 * 16 * 128 * 128;
  for (size_t i = (size_t)blockIdx.x * blockDim.x + threadIdx.x; i < total;
       i += (size_t)gridDim.x * blockDim.x) {
    size_t bc = i / 16384;               // b*16 + c
    int c  = (int)(bc % 16);
    int p  = (int)(i % 16384);
    int yo = p >> 7, xo = p & 127;
    const float* src = raw + bc * 65536;
    int o = (2 * yo) * 256 + 2 * xo;
    float v = 0.25f * (src[o] + src[o + 1] + src[o + 256] + src[o + 257]);
    v = (v - stats[2 * bc]) * stats[2 * bc + 1] * scale[c] + bias[c];
    act[i] = f2bf(v);
  }
}

// ---------------------------------------------------------------------------
// K4: implicit-GEMM conv via WMMA bf16 (fp32 accumulate) + TDM weight staging.
//   act [B][Cin][H][W] bf16    wgt [B][N][Kpad] bf16    raw [B][N][H][W] f32
//
// Workgroup = 128 threads (4 waves), owns one 16-pixel row tile.
//  1) wave 0 issues TENSOR_LOAD_TO_LDS: DMA the whole per-sample weight block
//     [N][Kpad] bf16 into LDS (overlaps with step 2).
//  2) all waves stage the 16 x Kpad im2col strip into LDS.
//  3) wave 0: s_wait_tensorcnt 0; barrier; each wave w runs the K-loop of
//     v_wmma_f32_16x16x32_bf16 for output-channel tile [16w, 16w+16).
//
// A frag (16-bit A 16x32, ISA 7.12.2): lane m=lane&15, h=lane>>4 reads
//   k in [kb+8h, kb+8h+8) then [kb+16+8h, +8)          -> two ds_load_b128.
// B frag (16-bit B, ISA 7.12.4 pattern): lane n=lane&15, h reads
//   k in [kb+8h, kb+8h+8) then [kb+16+8h, +8), as A                           -> two ds_load_b128.
// D: VGPR r <-> pixel x0 + 8h + r, lane&15 <-> out channel.
// ---------------------------------------------------------------------------
__global__ __launch_bounds__(128) void conv_wmma_kernel(
    const unsigned short* __restrict__ act,
    const unsigned short* __restrict__ wgt,
    float* __restrict__ raw,
    int Cin, int N, int H, int W, int K, int Kpad, int dil, int pad) {
  extern __shared__ unsigned short smA[];     // [16][Kpad+8]  then  [N][Kpad]
  const int pitch = Kpad + 8;                 // breaks LDS bank conflicts (A)
  const int b = blockIdx.y;
  const int tiles_per_row = W >> 4;
  const int y  = blockIdx.x / tiles_per_row;
  const int x0 = (blockIdx.x % tiles_per_row) << 4;
  const int HW = H * W;
  const int wave = threadIdx.x >> 5;
  const int lane = threadIdx.x & 31;
  unsigned short* smB = smA + 16 * pitch;

#if HAVE_TDM
  // ---- 1) async DMA of the weight block [N][Kpad] bf16 into LDS -----------
  if (wave == 0) {
    unsigned long long ga =
        (unsigned long long)(uintptr_t)(wgt + (size_t)b * N * Kpad);
    unsigned ldsoff = (unsigned)(16 * pitch) * 2u;       // byte offset of smB
    v4u g0;
    g0[0] = 1u;                                          // count=1, user mode
    g0[1] = ldsoff;                                      // lds_addr
    g0[2] = (unsigned)(ga & 0xFFFFFFFFu);                // global_addr lo
    g0[3] = (unsigned)((ga >> 32) & 0x01FFFFFFu) | (2u << 30);  // hi | type=2
    v8i g1;
    g1[0] = 0x00010000;                                  // data_size=1 (2B)
    g1[1] = (int)(((unsigned)Kpad & 0xFFFFu) << 16);     // tensor_dim0 lo
    g1[2] = (int)(((unsigned)Kpad >> 16) | ((unsigned)N << 16)); // |dim1 lo
    g1[3] = (int)(((unsigned)Kpad & 0xFFFFu) << 16);     // dim1 hi=0 |tile_dim0
    g1[4] = (int)(unsigned)N;                            // tile_dim1, tile_dim2=0
    g1[5] = (int)Kpad;                                   // tensor_dim0_stride lo
    g1[6] = 0;                                           // stride0 hi | stride1 lo
    g1[7] = 0;
    v4i gz4 = {0, 0, 0, 0};
    v8i gz8 = {0, 0, 0, 0, 0, 0, 0, 0};
    __builtin_amdgcn_tensor_load_to_lds(g0, g1, gz4, gz4, gz8, 0);
  }
#endif

  // ---- 2) stage im2col strip: smA[m][k] ------------------------------------
  const unsigned short* actb = act + (size_t)b * Cin * HW;
  for (int idx = threadIdx.x; idx < 16 * Kpad; idx += 128) {
    int m = idx / Kpad, k = idx - m * Kpad;
    unsigned short v = 0;
    if (k < K) {
      int ci = k / 25, r = k - ci * 25;
      int ky = r / 5,  kx = r - ky * 5;
      int iy = y + ky * dil - pad;
      int ix = x0 + m + kx * dil - pad;
      if ((unsigned)iy < (unsigned)H && (unsigned)ix < (unsigned)W)
        v = actb[(size_t)ci * HW + iy * W + ix];
    }
    smA[m * pitch + k] = v;
  }
#if HAVE_TDM
  if (wave == 0) __builtin_amdgcn_s_wait_tensorcnt(0);
#endif
  __syncthreads();

  const int NT = N >> 4;
  if (wave >= NT) return;

  const int n0 = wave << 4;
  const int lo = lane & 15;
  const int hi = lane >> 4;
  const unsigned short* arow = smA + lo * pitch;               // A: row = pixel
#if HAVE_TDM
  const unsigned short* wrow = smB + (size_t)(n0 + lo) * Kpad; // B from LDS
#else
  const unsigned short* wrow = wgt + ((size_t)b * N + (n0 + lo)) * Kpad;
#endif

  v8f acc = {};
  for (int kb = 0; kb < Kpad; kb += 32) {
#if !HAVE_TDM
    __builtin_prefetch(wrow + kb + 128, 0, 1);     // global_prefetch_b8
#endif
    u16x8 a0 = *(const u16x8*)(arow + kb + hi * 8);
    u16x8 a1 = *(const u16x8*)(arow + kb + 16 + hi * 8);
    u16x8 b0 = *(const u16x8*)(wrow + kb + hi * 8);
    u16x8 b1 = *(const u16x8*)(wrow + kb + 16 + hi * 8);
    u16x16 au = __builtin_shufflevector(a0, a1, 0,1,2,3,4,5,6,7,8,9,10,11,12,13,14,15);
    u16x16 bu = __builtin_shufflevector(b0, b1, 0,1,2,3,4,5,6,7,8,9,10,11,12,13,14,15);
    v16bf av = __builtin_bit_cast(v16bf, au);
    v16bf bv = __builtin_bit_cast(v16bf, bu);
    acc = __builtin_amdgcn_wmma_f32_16x16x32_bf16(
        /*neg_a=*/false, av, /*neg_b=*/false, bv,
        /*c_mod=*/(short)0, acc, /*reuse_a=*/false, /*reuse_b=*/false);
  }

  float* orow = raw + (((size_t)b * N + (n0 + lo)) * H + y) * W + x0 + hi * 8;
  ((float4*)orow)[0] = make_float4(acc[0], acc[1], acc[2], acc[3]);
  ((float4*)orow)[1] = make_float4(acc[4], acc[5], acc[6], acc[7]);
}

// ---------------------------------------------------------------------------
// K5: 1x1 conv, 64 -> 4 channels (tiny; plain VALU dot per pixel)
// ---------------------------------------------------------------------------
__global__ void conv7_kernel(const unsigned short* __restrict__ a6,
                             const unsigned short* __restrict__ w7,
                             float* __restrict__ r7) {
  size_t total = (size_t)16 * 16384;
  for (size_t i = (size_t)blockIdx.x * blockDim.x + threadIdx.x; i < total;
       i += (size_t)gridDim.x * blockDim.x) {
    int b = (int)(i / 16384), p = (int)(i % 16384);
    float a0 = 0.f, a1 = 0.f, a2 = 0.f, a3 = 0.f;
    const unsigned short* ab = a6 + (size_t)b * 64 * 16384 + p;
    const unsigned short* wb = w7 + (size_t)b * 4 * 64;
    for (int ci = 0; ci < 64; ++ci) {
      float v = bf2f(ab[(size_t)ci * 16384]);
      a0 += v * bf2f(wb[0 * 64 + ci]);
      a1 += v * bf2f(wb[1 * 64 + ci]);
      a2 += v * bf2f(wb[2 * 64 + ci]);
      a3 += v * bf2f(wb[3 * 64 + ci]);
    }
    float* ob = r7 + (size_t)b * 4 * 16384 + p;
    ob[0 * 16384] = a0; ob[1 * 16384] = a1; ob[2 * 16384] = a2; ob[3 * 16384] = a3;
  }
}

// ---------------------------------------------------------------------------
// K6: bilinear x2 upsample (align_corners=True, 128 -> 256) + residual add
// ---------------------------------------------------------------------------
__global__ void up_add_kernel(const float* __restrict__ r7,
                              const float* __restrict__ x,
                              float* __restrict__ out) {
  size_t total = (size_t)16 * 4 * 256 * 256;
  const float scl = 127.f / 255.f;
  for (size_t i = (size_t)blockIdx.x * blockDim.x + threadIdx.x; i < total;
       i += (size_t)gridDim.x * blockDim.x) {
    size_t bc = i / 65536;               // b*4 + c
    int p = (int)(i % 65536);
    int Y = p >> 8, X = p & 255;
    float py = Y * scl, px = X * scl;
    int y0 = (int)py, xx0 = (int)px;
    float fy = py - y0, fx = px - xx0;
    int y1 = min(y0 + 1, 127), x1 = min(xx0 + 1, 127);
    const float* src = r7 + bc * 16384;
    float v00 = src[y0 * 128 + xx0], v01 = src[y0 * 128 + x1];
    float v10 = src[y1 * 128 + xx0], v11 = src[y1 * 128 + x1];
    float v = (v00 * (1.f - fx) + v01 * fx) * (1.f - fy) +
              (v10 * (1.f - fx) + v11 * fx) * fy;
    out[i] = v + x[i];
  }
}

// ---------------------------------------------------------------------------
// Host orchestration
// ---------------------------------------------------------------------------
struct LayerCfg { int N, K, Kpad; };
static const LayerCfg LW[7] = {
    {16, 100,  128}, {16, 400, 416}, {32, 400, 416}, {32, 800, 800},
    {64, 800,  800}, {64, 1600, 1600}, {4, 64, 64}};

static inline size_t alignup(size_t v) { return (v + 255) & ~(size_t)255; }

extern "C" void kernel_launch(void* const* d_in, const int* in_sizes, int n_in,
                              void* d_out, int out_size, void* d_ws, size_t ws_size,
                              hipStream_t stream) {
  (void)in_sizes; (void)n_in; (void)out_size; (void)ws_size;
  const float* x      = (const float*)d_in[0];
  const int*   action = (const int*)d_in[1];
  const float* bnS[7], *bnB[7];
  for (int i = 0; i < 7; ++i) { bnS[i] = (const float*)d_in[16 + 2 * i];
                                bnB[i] = (const float*)d_in[17 + 2 * i]; }

  // ---- workspace layout ----------------------------------------------------
  char* base = (char*)d_ws;
  size_t off = 0;
  unsigned short* wgt[7];
  for (int l = 0; l < 7; ++l) {
    wgt[l] = (unsigned short*)(base + off);
    off = alignup(off + (size_t)16 * LW[l].N * LW[l].Kpad * 2);
  }
  float* stats = (float*)(base + off);  off = alignup(off + 16 * 64 * 2 * 4);
  unsigned short* xn0 = (unsigned short*)(base + off); off = alignup(off + (size_t)16*4*65536*2);
  float* raw = (float*)(base + off);    off = alignup(off + (size_t)16*16*65536*4);
  unsigned short* a1 = (unsigned short*)(base + off);  off = alignup(off + (size_t)16*16*65536*2);
  unsigned short* a2 = (unsigned short*)(base + off);  off = alignup(off + (size_t)16*16*16384*2);
  unsigned short* a3 = (unsigned short*)(base + off);  off = alignup(off + (size_t)16*32*16384*2);
  unsigned short* a4 = (unsigned short*)(base + off);  off = alignup(off + (size_t)16*32*16384*2);
  unsigned short* a5 = a1;  // a1 is dead once conv2's raw output exists
  unsigned short* a6 = (unsigned short*)(base + off);  off = alignup(off + (size_t)16*64*16384*2);
  float* r7 = (float*)(base + off);     off = alignup(off + (size_t)16*4*16384*4);

  // ---- 0) gather per-sample conv weights (one-hot matmul == column pick) ---
  for (int l = 0; l < 7; ++l) {
    size_t tot = (size_t)16 * LW[l].N * LW[l].Kpad;
    int grid = (int)((tot + 255) / 256);
    gather_w_kernel<<<grid, 256, 0, stream>>>(
        (const float*)d_in[2 + 2 * l], (const float*)d_in[3 + 2 * l],
        action, wgt[l], LW[l].N, LW[l].K, LW[l].Kpad);
  }

  auto stats_apply = [&](const float* src, unsigned short* dst, int ch, int hw,
                         int layer, int relu) {
    bn_stats_kernel<<<16 * ch, 256, 0, stream>>>(src, stats, hw);
    size_t tot = (size_t)16 * ch * hw;
    bn_apply_kernel<<<(int)((tot + 255) / 256), 256, 0, stream>>>(
        src, dst, stats, bnS[layer], bnB[layer], ch, hw, relu, tot);
  };
  auto conv = [&](const unsigned short* act, int layer, int cin, int hw_side,
                  int dil, int pad) {
    const LayerCfg& L = LW[layer];
    dim3 grid(hw_side * hw_side / 16, 16, 1);
    size_t shmem = (size_t)16 * (L.Kpad + 8) * 2      // im2col A strip
                 + (size_t)L.N * L.Kpad * 2;          // TDM-staged weights
    conv_wmma_kernel<<<grid, 128, shmem, stream>>>(
        act, wgt[layer], raw, cin, L.N, hw_side, hw_side, L.K, L.Kpad, dil, pad);
  };

  // ---- pipeline ------------------------------------------------------------
  stats_apply(x, xn0, 4, 65536, 0, 0);                 // bn0(x) -> bf16
  conv(xn0, 0, 4, 256, 2, 4);                          // conv1 (dil2,pad4)
  stats_apply(raw, a1, 16, 65536, 1, 1);               // relu(bn1)
  conv(a1, 1, 16, 256, 1, 2);                          // conv2 (pad2)
  bn_stats_kernel<<<16 * 16, 256, 0, stream>>>(raw, stats, 65536);
  {                                                    // avgpool2x2 + bn2
    size_t tot = (size_t)16 * 16 * 16384;
    pool_bn_kernel<<<(int)((tot + 255) / 256), 256, 0, stream>>>(
        raw, a2, stats, bnS[2], bnB[2]);
  }
  conv(a2, 2, 16, 128, 2, 4);                          // conv3
  stats_apply(raw, a3, 32, 16384, 3, 1);               // relu(bn3)
  conv(a3, 3, 32, 128, 1, 2);                          // conv4
  stats_apply(raw, a4, 32, 16384, 4, 0);               // bn4
  conv(a4, 4, 32, 128, 2, 4);                          // conv5
  stats_apply(raw, a5, 64, 16384, 5, 1);               // relu(bn5)
  conv(a5, 5, 64, 128, 1, 2);                          // conv6
  stats_apply(raw, a6, 64, 16384, 6, 0);               // bn6
  {
    size_t tot = (size_t)16 * 16384;
    conv7_kernel<<<(int)((tot + 255) / 256), 256, 0, stream>>>(a6, wgt[6], r7);
  }
  {
    size_t tot = (size_t)16 * 4 * 65536;
    up_add_kernel<<<(int)((tot + 255) / 256), 256, 0, stream>>>(
        r7, x, (float*)d_out);
  }
}
